// Hebbian_Block_61607010894203
// MI455X (gfx1250) — matrix operand order repaired and checked by reading, not compiled, not run on hardware
//
#include <hip/hip_runtime.h>
#include <stdint.h>

#define S_LEN   4096
#define D_MODEL 512
#define N_HEADS 8
#define HEAD_DIM 64
#define LN_EPS  1e-5f

typedef __attribute__((ext_vector_type(16))) __bf16 v16bf;
typedef __attribute__((ext_vector_type(2)))  __bf16 v2bf;
typedef __attribute__((ext_vector_type(8)))  float  v8f;

union BfTile {
    v16bf v;
    unsigned short u[16];
    uint4 q[2];
};

__device__ __forceinline__ unsigned short f2bf(float f) {
    union { float f; uint32_t u; } c; c.f = f;
    uint32_t u = c.u;
    uint32_t r = u + 0x7FFFu + ((u >> 16) & 1u);   // round-to-nearest-even
    return (unsigned short)(r >> 16);
}

// pack two floats into {bf16(lo), bf16(hi)} in one dword
__device__ __forceinline__ uint32_t pack2bf(float lo, float hi) {
#if __has_builtin(__builtin_amdgcn_cvt_pk_bf16_f32)
    union { v2bf v; uint32_t u; } c;
    c.v = __builtin_amdgcn_cvt_pk_bf16_f32(lo, hi);
    return c.u;
#else
    return (uint32_t)f2bf(lo) | ((uint32_t)f2bf(hi) << 16);
#endif
}

__device__ __forceinline__ v8f wmma_bf16(v16bf a, v16bf b, v8f c) {
    return __builtin_amdgcn_wmma_f32_16x16x32_bf16(false, a, false, b, (short)0, c, false, false);
}

// A-tile (16x32 bf16, M x K): lane holds row M = lane&15.
__device__ __forceinline__ v16bf load_a_tile(const unsigned short* base, int ld, int lane) {
    BfTile t;
    const unsigned short* p = base + (lane & 15) * ld + ((lane >> 4) << 3);
    t.q[0] = *(const uint4*)(p);
    t.q[1] = *(const uint4*)(p + 16);
    return t.v;
}

// B-tile (32x16 bf16, K x N) read from an [N][K]-major source (B = src^T):
__device__ __forceinline__ v16bf load_b_rows(const unsigned short* base, int ld, int lane) {
    BfTile t;
    const unsigned short* p = base + (lane & 15) * ld + ((lane >> 4) << 3);
    t.q[0] = *(const uint4*)(p);
    t.q[1] = *(const uint4*)(p + 16);
    return t.v;
}

#define V8F_ZERO {0.f,0.f,0.f,0.f,0.f,0.f,0.f,0.f}

// ---------------------------------------------------------------- LayerNorm
__global__ void ln_kernel(const float* __restrict__ x, const float* __restrict__ g,
                          const float* __restrict__ b, unsigned short* __restrict__ h) {
    int wave = (blockIdx.x * blockDim.x + threadIdx.x) >> 5;
    int lane = threadIdx.x & 31;
    const float* row = x + wave * D_MODEL;
    float vals[16];
    float sum = 0.f;
#pragma unroll
    for (int i = 0; i < 16; ++i) { vals[i] = row[lane + 32 * i]; sum += vals[i]; }
#pragma unroll
    for (int m = 16; m >= 1; m >>= 1) sum += __shfl_xor(sum, m, 32);
    float mu = sum * (1.f / D_MODEL);
    float vs = 0.f;
#pragma unroll
    for (int i = 0; i < 16; ++i) { float d = vals[i] - mu; vs += d * d; }
#pragma unroll
    for (int m = 16; m >= 1; m >>= 1) vs += __shfl_xor(vs, m, 32);
    float inv = rsqrtf(vs * (1.f / D_MODEL) + LN_EPS);
#pragma unroll
    for (int i = 0; i < 16; ++i) {
        int c = lane + 32 * i;
        h[wave * D_MODEL + c] = f2bf((vals[i] - mu) * inv * g[c] + b[c]);
    }
}

// ------------------------------------------- weight convert f32[in][out] -> bf16[out][in]
__global__ void wconv_kernel(const float* __restrict__ w, unsigned short* __restrict__ wt) {
    int idx = blockIdx.x * blockDim.x + threadIdx.x;   // 512*512 exact
    int i = idx >> 9, o = idx & 511;
    wt[o * D_MODEL + i] = f2bf(w[idx]);
}

// ------------------------------------------------------------- QKV projection GEMM
// one wave = one (matrix, 32-row tile, head); 384 blocks x 8 waves = 3072
// Q,K stored head-major [H][S][64].
// V stored transposed [H][64][S] with keys INTERLEAVED within each 32-row block:
// s_local -> 2*(r+8*rh)+g, matching the packed-pair LDS staging order in attention.
__global__ void qkv_gemm_kernel(const unsigned short* __restrict__ h,
                                const unsigned short* __restrict__ wtq,
                                const unsigned short* __restrict__ wtk,
                                const unsigned short* __restrict__ wtv,
                                const float* __restrict__ bq, const float* __restrict__ bk,
                                const float* __restrict__ bv,
                                unsigned short* __restrict__ qh, unsigned short* __restrict__ kh,
                                unsigned short* __restrict__ vt) {
    int gw = (blockIdx.x * blockDim.x + threadIdx.x) >> 5;
    int lane = threadIdx.x & 31;
    int mat = gw / 1024, rem = gw % 1024;
    int rt = rem >> 3, ct = rem & 7;     // ct == head
    const unsigned short* wt = (mat == 0) ? wtq : (mat == 1) ? wtk : wtv;
    const float* bias        = (mat == 0) ? bq  : (mat == 1) ? bk  : bv;

    v8f acc[2][4] = {{V8F_ZERO, V8F_ZERO, V8F_ZERO, V8F_ZERO},
                     {V8F_ZERO, V8F_ZERO, V8F_ZERO, V8F_ZERO}};
    for (int k0 = 0; k0 < D_MODEL; k0 += 32) {
        v16bf a0 = load_a_tile(h + (rt * 32 + 0)  * D_MODEL + k0, D_MODEL, lane);
        v16bf a1 = load_a_tile(h + (rt * 32 + 16) * D_MODEL + k0, D_MODEL, lane);
#pragma unroll
        for (int t = 0; t < 4; ++t) {
            v16bf b = load_b_rows(wt + (ct * 64 + t * 16) * D_MODEL + k0, D_MODEL, lane);
            acc[0][t] = wmma_bf16(a0, b, acc[0][t]);
            acc[1][t] = wmma_bf16(a1, b, acc[1][t]);
        }
    }
    int col = lane & 15, rh = lane >> 4;
    if (mat == 2) {
        // V: store transposed + key-interleaved: [H][64][S], pos = rt*32 + 2*(r+8*rh)+g
#pragma unroll
        for (int g = 0; g < 2; ++g)
#pragma unroll
            for (int t = 0; t < 4; ++t) {
                int d = t * 16 + col;
                float bval = bias[ct * 64 + d];
#pragma unroll
                for (int r = 0; r < 8; ++r) {
                    int pos = rt * 32 + 2 * (r + 8 * rh) + g;
                    vt[(size_t)ct * HEAD_DIM * S_LEN + (size_t)d * S_LEN + pos] =
                        f2bf(acc[g][t][r] + bval);
                }
            }
    } else {
        unsigned short* dst = (mat == 0) ? qh : kh;
#pragma unroll
        for (int g = 0; g < 2; ++g)
#pragma unroll
            for (int t = 0; t < 4; ++t) {
                int d = t * 16 + col;
                float bval = bias[ct * 64 + d];
#pragma unroll
                for (int r = 0; r < 8; ++r) {
                    int row = rt * 32 + g * 16 + r + 8 * rh;
                    dst[(size_t)ct * S_LEN * HEAD_DIM + row * HEAD_DIM + d] =
                        f2bf(acc[g][t][r] + bval);
                }
            }
    }
}

// ------------------------------------------------------------------- attention
// one 32x32 key-block step; MASKED only for the diagonal block.
template <bool MASKED>
__device__ __forceinline__ void attn_block(const v16bf aq[2][2], v8f o[2][4],
                                           unsigned short* myaff,
                                           const unsigned short* __restrict__ K,
                                           const unsigned short* __restrict__ Vt,
                                           int jb, int qbase, float beta,
                                           int lane, int col, int rh) {
    // QK^T : 4 B-tiles shared by 2 query groups
    v16bf bk0 = load_b_rows(K + (jb)      * HEAD_DIM + 0,  HEAD_DIM, lane);
    v16bf bk1 = load_b_rows(K + (jb)      * HEAD_DIM + 32, HEAD_DIM, lane);
    v16bf bk2 = load_b_rows(K + (jb + 16) * HEAD_DIM + 0,  HEAD_DIM, lane);
    v16bf bk3 = load_b_rows(K + (jb + 16) * HEAD_DIM + 32, HEAD_DIM, lane);
    uint32_t* aff32 = (uint32_t*)myaff;
#pragma unroll
    for (int g = 0; g < 2; ++g) {
        v8f c0 = V8F_ZERO, c1 = V8F_ZERO;
        c0 = wmma_bf16(aq[g][0], bk0, c0);
        c0 = wmma_bf16(aq[g][1], bk1, c0);
        c1 = wmma_bf16(aq[g][0], bk2, c1);
        c1 = wmma_bf16(aq[g][1], bk3, c1);
        // beta-scale + relu (+ causal mask only on the diagonal block);
        // packed pair (key col, key col+16) -> interleaved K positions (2*col, 2*col+1)
#pragma unroll
        for (int r = 0; r < 8; ++r) {
            float v0 = fmaxf(c0[r] * beta, 0.f);
            float v1 = fmaxf(c1[r] * beta, 0.f);
            if (MASKED) {
                int m = qbase + g * 16 + r + 8 * rh;
                if (jb + col > m)      v0 = 0.f;
                if (jb + 16 + col > m) v1 = 0.f;
            }
            aff32[(g * 16 + r + 8 * rh) * 16 + col] = pack2bf(v0, v1);
        }
    }
    v16bf a0 = load_a_tile(myaff,           32, lane);   // ds_load_b128 x2
    v16bf a1 = load_a_tile(myaff + 16 * 32, 32, lane);
    // out += aff(32x32) @ V(32x64); Vt is key-interleaved to match A's K order
#pragma unroll
    for (int t = 0; t < 4; ++t) {
        v16bf bvt = load_b_rows(Vt + (size_t)(t * 16) * S_LEN + jb, S_LEN, lane);
        o[0][t] = wmma_bf16(a0, bvt, o[0][t]);
        o[1][t] = wmma_bf16(a1, bvt, o[1][t]);
    }
}

// one wave = one (32-query tile, head); 128 blocks x 8 waves = 1024
__global__ void attn_kernel(const unsigned short* __restrict__ qh,
                            const unsigned short* __restrict__ kh,
                            const unsigned short* __restrict__ vt,
                            const float* __restrict__ beta_p,
                            unsigned short* __restrict__ attn) {
    __shared__ unsigned short lds_aff[8][32 * 32];   // per-wave 32x32 bf16 staging (16 KB)
    int wib  = threadIdx.x >> 5;
    int lane = threadIdx.x & 31;
    int gw   = blockIdx.x * 8 + wib;
    int qt = gw >> 3, head = gw & 7;
    int qbase = qt * 32;
    float beta = beta_p[0];

    const unsigned short* Q  = qh + (size_t)head * S_LEN * HEAD_DIM;   // [S][64]
    const unsigned short* K  = kh + (size_t)head * S_LEN * HEAD_DIM;   // [S][64]
    const unsigned short* Vt = vt + (size_t)head * HEAD_DIM * S_LEN;   // [64][S] interleaved

    v16bf aq[2][2];
#pragma unroll
    for (int g = 0; g < 2; ++g)
#pragma unroll
        for (int hh = 0; hh < 2; ++hh)
            aq[g][hh] = load_a_tile(Q + (qbase + g * 16) * HEAD_DIM + hh * 32, HEAD_DIM, lane);

    v8f o[2][4] = {{V8F_ZERO, V8F_ZERO, V8F_ZERO, V8F_ZERO},
                   {V8F_ZERO, V8F_ZERO, V8F_ZERO, V8F_ZERO}};
    int col = lane & 15, rh = lane >> 4;
    unsigned short* myaff = &lds_aff[wib][0];

    // fully-causal key blocks: no per-element mask needed
    for (int jb = 0; jb < qbase; jb += 32) {
        __builtin_prefetch(K + (jb + 32) * HEAD_DIM, 0, 1);
        attn_block<false>(aq, o, myaff, K, Vt, jb, qbase, beta, lane, col, rh);
    }
    // diagonal block (jb == qbase): masked
    attn_block<true>(aq, o, myaff, K, Vt, qbase, qbase, beta, lane, col, rh);

    // store head-interleaved bf16 [S][512]
#pragma unroll
    for (int g = 0; g < 2; ++g)
#pragma unroll
        for (int t = 0; t < 4; ++t) {
            int d = t * 16 + col;
#pragma unroll
            for (int r = 0; r < 8; ++r) {
                int row = qbase + g * 16 + r + 8 * rh;
                attn[(size_t)row * D_MODEL + head * 64 + d] = f2bf(o[g][t][r]);
            }
        }
}

// --------------------------------------------------- output projection + residual
// one wave = one 32x64 tile; 128 blocks x 8 waves = 1024
__global__ void out_gemm_kernel(const unsigned short* __restrict__ attn,
                                const unsigned short* __restrict__ wto,
                                const float* __restrict__ bo, const float* __restrict__ x,
                                float* __restrict__ out) {
    int gw = (blockIdx.x * blockDim.x + threadIdx.x) >> 5;
    int lane = threadIdx.x & 31;
    int rt = gw >> 3, ct = gw & 7;
    v8f acc[2][4] = {{V8F_ZERO, V8F_ZERO, V8F_ZERO, V8F_ZERO},
                     {V8F_ZERO, V8F_ZERO, V8F_ZERO, V8F_ZERO}};
    for (int k0 = 0; k0 < D_MODEL; k0 += 32) {
        v16bf a0 = load_a_tile(attn + (rt * 32 + 0)  * D_MODEL + k0, D_MODEL, lane);
        v16bf a1 = load_a_tile(attn + (rt * 32 + 16) * D_MODEL + k0, D_MODEL, lane);
#pragma unroll
        for (int t = 0; t < 4; ++t) {
            v16bf b = load_b_rows(wto + (ct * 64 + t * 16) * D_MODEL + k0, D_MODEL, lane);
            acc[0][t] = wmma_bf16(a0, b, acc[0][t]);
            acc[1][t] = wmma_bf16(a1, b, acc[1][t]);
        }
    }
    int col = lane & 15, rh = lane >> 4;
#pragma unroll
    for (int g = 0; g < 2; ++g)
#pragma unroll
        for (int t = 0; t < 4; ++t) {
            int cg = ct * 64 + t * 16 + col;
            float bval = bo[cg];
#pragma unroll
            for (int r = 0; r < 8; ++r) {
                int row = rt * 32 + g * 16 + r + 8 * rh;
                out[(size_t)row * D_MODEL + cg] = x[(size_t)row * D_MODEL + cg] + acc[g][t][r] + bval;
            }
        }
}

extern "C" void kernel_launch(void* const* d_in, const int* in_sizes, int n_in,
                              void* d_out, int out_size, void* d_ws, size_t ws_size,
                              hipStream_t stream) {
    (void)in_sizes; (void)n_in; (void)out_size; (void)ws_size;
    const float* x    = (const float*)d_in[0];
    const float* wq   = (const float*)d_in[1];
    const float* bq   = (const float*)d_in[2];
    const float* wk   = (const float*)d_in[3];
    const float* bk   = (const float*)d_in[4];
    const float* wv   = (const float*)d_in[5];
    const float* bv   = (const float*)d_in[6];
    const float* wo   = (const float*)d_in[7];
    const float* bo   = (const float*)d_in[8];
    const float* ln_g = (const float*)d_in[9];
    const float* ln_b = (const float*)d_in[10];
    const float* beta = (const float*)d_in[11];

    char* ws = (char*)d_ws;
    unsigned short* h    = (unsigned short*)(ws);                     // 4 MB
    unsigned short* wtq  = (unsigned short*)(ws + (4u  << 20));       // 0.5 MB each
    unsigned short* wtk  = wtq + D_MODEL * D_MODEL;
    unsigned short* wtv  = wtk + D_MODEL * D_MODEL;
    unsigned short* wto  = wtv + D_MODEL * D_MODEL;                   // ends at 6 MB
    unsigned short* qh   = (unsigned short*)(ws + (6u  << 20));       // 4 MB, [H][S][64]
    unsigned short* kh   = (unsigned short*)(ws + (10u << 20));       // 4 MB, [H][S][64]
    unsigned short* vt   = (unsigned short*)(ws + (14u << 20));       // 4 MB, [H][64][S]
    unsigned short* attn = (unsigned short*)(ws + (18u << 20));       // 4 MB -> 22 MB total

    ln_kernel<<<512, 256, 0, stream>>>(x, ln_g, ln_b, h);
    wconv_kernel<<<1024, 256, 0, stream>>>(wq, wtq);
    wconv_kernel<<<1024, 256, 0, stream>>>(wk, wtk);
    wconv_kernel<<<1024, 256, 0, stream>>>(wv, wtv);
    wconv_kernel<<<1024, 256, 0, stream>>>(wo, wto);
    qkv_gemm_kernel<<<384, 256, 0, stream>>>(h, wtq, wtk, wtv, bq, bk, bv, qh, kh, vt);
    attn_kernel<<<128, 256, 0, stream>>>(qh, kh, vt, beta, attn);
    out_gemm_kernel<<<128, 256, 0, stream>>>(attn, wto, bo, x, (float*)d_out);
}
